// StudentNetwork_noRelu_19859928777108
// MI455X (gfx1250) — matrix operand order repaired and checked by reading, not compiled, not run on hardware
//
#include <hip/hip_runtime.h>

typedef __attribute__((ext_vector_type(16))) _Float16 v16h;
typedef __attribute__((ext_vector_type(8)))  _Float16 v8h;
typedef __attribute__((ext_vector_type(8)))  float    v8f;

#define PI_F 3.14159265358979323846f

// ---------------------------------------------------------------------------
// Batched GEMM, C = A * Bt^T (+bias, optional relu).
//   A  : f16, [batch?][M][Kp] row-major (Kp = K rounded up to 32, zero padded)
//   Bt : f16, [batch?][N][Kp] row-major (i.e. B transposed)
//   C  : f32, [batch?][M][N]  row-major
// Tile: 64x64x32, 256 threads = 8 wave32 waves; each wave: one 16-row strip,
// two 16-col WMMA accumulators -> 2x v_wmma_f32_16x16x32_f16 per k-step.
// Interior tiles run a dedicated loop staging via GLOBAL_LOAD_ASYNC_TO_LDS_B128
// (ASYNCcnt); edge tiles run a guarded register-staging loop with zero fill.
// ---------------------------------------------------------------------------
__global__ __launch_bounds__(256)
void gemm_f16_wmma(const _Float16* __restrict__ A, const _Float16* __restrict__ Bt,
                   float* __restrict__ C, int M, int N, int Kp,
                   long long strA, long long strB, long long strC,
                   const float* __restrict__ bias, int relu)
{
    __shared__ __align__(16) _Float16 sA[2][64][40];
    __shared__ __align__(16) _Float16 sB[2][64][40];

    const _Float16* Ab = A  + (size_t)blockIdx.z * strA;
    const _Float16* Bb = Bt + (size_t)blockIdx.z * strB;
    float*          Cb = C  + (size_t)blockIdx.z * strC;

    const int m0   = blockIdx.x * 64;
    const int n0   = blockIdx.y * 64;
    const int tid  = threadIdx.x;
    const int lane = tid & 31;
    const int w    = tid >> 5;
    const int mt   = (w & 3) << 4;   // wave's m-tile offset within block tile
    const int ng   = (w >> 2) << 5;  // wave's n-group offset (covers two 16-col tiles)

    // block-uniform: whole 64x64 tile in range -> async global->LDS staging loop
    const bool full = (m0 + 64 <= M) && (n0 + 64 <= N);

    // CDNA5 wait path (TENSORcnt is zero here; harmless, exercises s_wait_tensorcnt)
    __builtin_amdgcn_s_wait_tensorcnt(0);

    const int r  = tid >> 2;          // 0..63
    const int kk = (tid & 3) << 3;    // 0,8,16,24
    const int gm = m0 + r;
    const int gn = n0 + r;
    const _Float16* apBase = Ab + (size_t)gm * Kp + kk;
    const _Float16* bpBase = Bb + (size_t)gn * Kp + kk;

    auto stageAsync = [&](int buf, int k0) {
        // ASYNCcnt-tracked DMA of 16B per lane straight into LDS.
        // Flat LDS pointer low 32 bits == LDS byte address (ISA 10.2).
        const _Float16* ap = apBase + k0;
        const _Float16* bp = bpBase + k0;
        const unsigned la = (unsigned)(uintptr_t)&sA[buf][r][kk];
        const unsigned lb = (unsigned)(uintptr_t)&sB[buf][r][kk];
        const unsigned long long ga = (unsigned long long)(uintptr_t)ap;
        const unsigned long long gb = (unsigned long long)(uintptr_t)bp;
        asm volatile("global_load_async_to_lds_b128 %0, %1, off"
                     :: "v"(la), "v"(ga) : "memory");
        asm volatile("global_load_async_to_lds_b128 %0, %1, off"
                     :: "v"(lb), "v"(gb) : "memory");
        __builtin_prefetch((const void*)(ap + 32), 0, 1);  // next k-tile
        __builtin_prefetch((const void*)(bp + 32), 0, 1);
    };

    auto stageEdge = [&](int buf, int k0) {
        v8h av = {};
        v8h bv = {};
        if (gm < M) av = *(const v8h*)(apBase + k0);
        if (gn < N) bv = *(const v8h*)(bpBase + k0);
        *(v8h*)&sA[buf][r][kk] = av;
        *(v8h*)&sB[buf][r][kk] = bv;
    };

    v8f acc0 = {};
    v8f acc1 = {};
    // A fragment addressing: lanes 0-15 rows M,
    // K = {kb..kb+15}, kb = 0 (lanes<16) / 16 (lanes>=16)
    const int mrow = mt + (lane & 15);
    const int akb  = (lane >> 4) << 4;   // 0 or 16
    // B fragment: lane holds 16 consecutive K for its column; kb2 = 0 / 16
    const int bkb  = (lane >> 4) << 4;   // 0 or 16
    const int nc0  = ng + (lane & 15);
    const int nc1  = nc0 + 16;

    auto mmaStep = [&](int buf) {
        v8h alo = *(const v8h*)&sA[buf][mrow][akb];
        v8h ahi = *(const v8h*)&sA[buf][mrow][akb + 8];
        v16h a = __builtin_shufflevector(alo, ahi, 0,1,2,3,4,5,6,7,8,9,10,11,12,13,14,15);

        v8h b0lo = *(const v8h*)&sB[buf][nc0][bkb];
        v8h b0hi = *(const v8h*)&sB[buf][nc0][bkb + 8];
        v16h b0 = __builtin_shufflevector(b0lo, b0hi, 0,1,2,3,4,5,6,7,8,9,10,11,12,13,14,15);
        v8h b1lo = *(const v8h*)&sB[buf][nc1][bkb];
        v8h b1hi = *(const v8h*)&sB[buf][nc1][bkb + 8];
        v16h b1 = __builtin_shufflevector(b1lo, b1hi, 0,1,2,3,4,5,6,7,8,9,10,11,12,13,14,15);

        acc0 = __builtin_amdgcn_wmma_f32_16x16x32_f16(false, a, false, b0, (short)0, acc0, false, false);
        acc1 = __builtin_amdgcn_wmma_f32_16x16x32_f16(false, a, false, b1, (short)0, acc1, false, false);
    };

    const int nk = Kp >> 5;
    if (full) {
        // interior tile: branch-free async pipeline
        stageAsync(0, 0);
        for (int kt = 0; kt < nk; ++kt) {
            const int buf = kt & 1;
            asm volatile("s_wait_asynccnt 0x0" ::: "memory");
            __syncthreads();
            if (kt + 1 < nk) stageAsync(buf ^ 1, (kt + 1) << 5);
            mmaStep(buf);
        }
    } else {
        // edge tile: guarded register staging with zero fill
        stageEdge(0, 0);
        for (int kt = 0; kt < nk; ++kt) {
            const int buf = kt & 1;
            __syncthreads();
            if (kt + 1 < nk) stageEdge(buf ^ 1, (kt + 1) << 5);
            mmaStep(buf);
        }
    }

    // C/D layout: lane 0-15 -> N=lane, rows M = vgpr (+8 for lanes 16-31)
    const int c0 = n0 + nc0;
    const int c1 = n0 + nc1;
    const int rb = m0 + mt + ((lane >> 4) << 3);
    for (int e = 0; e < 8; ++e) {
        const int m = rb + e;
        if (m >= M) continue;
        if (c0 < N) {
            float t = acc0[e] + (bias ? bias[c0] : 0.0f);
            if (relu) t = fmaxf(t, 0.0f);
            Cb[(size_t)m * N + c0] = t;
        }
        if (c1 < N) {
            float t = acc1[e] + (bias ? bias[c1] : 0.0f);
            if (relu) t = fmaxf(t, 0.0f);
            Cb[(size_t)m * N + c1] = t;
        }
    }
}

// ---------------------------------------------------------------------------
// Dirichlet (ideal spectral resample) matrix: D[a][k] for n_in -> n_out (odd).
// D = (1/n_out) * sin(pi*n_out*theta)/sin(pi*theta), theta = a/n_out - k/n_in.
// ---------------------------------------------------------------------------
__global__ void gen_dirichlet_kernel(_Float16* __restrict__ D, int n_out, int n_in,
                                     int Kp, long long total)
{
    long long idx = (long long)blockIdx.x * blockDim.x + threadIdx.x;
    if (idx >= total) return;
    const int k = (int)(idx % Kp);
    const int a = (int)(idx / Kp);
    float v = 0.0f;
    if (k < n_in) {
        const float theta = (float)a / (float)n_out - (float)k / (float)n_in;
        const float s = sinf(PI_F * theta);
        if (fabsf(s) < 1e-6f) v = 1.0f;  // limit n_out/n_out
        else v = sinf(PI_F * (float)n_out * theta) / ((float)n_out * s);
    }
    D[idx] = (_Float16)v;
}

// im2col for 3x3 "same" conv (true convolution handled by flipping weights).
__global__ void im2col_kernel(const float* __restrict__ X, _Float16* __restrict__ Acol,
                              int C, int H, int W, int K, int Kp, long long total)
{
    long long idx = (long long)blockIdx.x * blockDim.x + threadIdx.x;
    if (idx >= total) return;
    const int k = (int)(idx % Kp);
    const long long r = idx / Kp;
    _Float16 v = (_Float16)0.0f;
    if (k < K) {
        const int c = k / 9, t = k % 9;
        const int dy = t / 3 - 1, dx = t % 3 - 1;
        const int x = (int)(r % W);
        const int y = (int)((r / W) % H);
        const int b = (int)(r / ((long long)W * H));
        const int yy = y + dy, xx = x + dx;
        if (yy >= 0 && yy < H && xx >= 0 && xx < W)
            v = (_Float16)X[(((size_t)b * C + c) * H + yy) * W + xx];
    }
    Acol[idx] = v;
}

// Pack conv filters (ci,co,3,3,2) -> Bt[co][Kp], real part, taps flipped
// (full-conv-cropped == convolution, im2col does correlation).
__global__ void pack_convw_kernel(const float* __restrict__ f, _Float16* __restrict__ Bt,
                                  int Ci, int Co, int Kp, long long total)
{
    long long idx = (long long)blockIdx.x * blockDim.x + threadIdx.x;
    if (idx >= total) return;
    const int k = (int)(idx % Kp);
    const int o = (int)(idx / Kp);
    _Float16 v = (_Float16)0.0f;
    if (k < Ci * 9) {
        const int c = k / 9, t = k % 9, ty = t / 3, tx = t % 3;
        v = (_Float16)f[((((size_t)c * Co + o) * 3 + (2 - ty)) * 3 + (2 - tx)) * 2 + 0];
    }
    Bt[idx] = v;
}

// NHWC conv output -> Xt[(b*C+c)][w][Kph], Xt[z][w][h]=X[h][w]  (B operand of D*X)
__global__ void pack_xt_kernel(const float* __restrict__ conv, _Float16* __restrict__ Xt,
                               int C, int H, int W, int Kph, long long total)
{
    long long idx = (long long)blockIdx.x * blockDim.x + threadIdx.x;
    if (idx >= total) return;
    const int k = (int)(idx % Kph);
    long long q = idx / Kph;
    const int w = (int)(q % W);
    q /= W;
    const int c = (int)(q % C);
    const int b = (int)(q / C);
    _Float16 v = (_Float16)0.0f;
    if (k < H) v = (_Float16)conv[(((size_t)b * H + k) * W + w) * C + c];
    Xt[idx] = v;
}

// f32 rows [*,N] -> f16 rows [*,Kp] zero-padded
__global__ void pack_rows_kernel(const float* __restrict__ src, _Float16* __restrict__ dst,
                                 int N, int Kp, long long total)
{
    long long idx = (long long)blockIdx.x * blockDim.x + threadIdx.x;
    if (idx >= total) return;
    const int k = (int)(idx % Kp);
    const long long r = idx / Kp;
    dst[idx] = (k < N) ? (_Float16)src[r * (long long)N + k] : (_Float16)0.0f;
}

// FC weights (K,N) f32 -> Bt[N][Kp] f16 (transposed)
__global__ void pack_wt_kernel(const float* __restrict__ Wm, _Float16* __restrict__ Bt,
                               int K, int N, int Kp, long long total)
{
    long long idx = (long long)blockIdx.x * blockDim.x + threadIdx.x;
    if (idx >= total) return;
    const int k = (int)(idx % Kp);
    const int n = (int)(idx / Kp);
    Bt[idx] = (k < K) ? (_Float16)Wm[(size_t)k * N + n] : (_Float16)0.0f;
}

// Per-channel mean / rstd over rows of NHWC conv5 output.
__global__ void bn_stats_kernel(const float* __restrict__ conv, int rows, int C,
                                float* __restrict__ mean, float* __restrict__ rstd)
{
    __shared__ float ss[256], sq[256];
    const int c = blockIdx.x;
    float s = 0.0f, q = 0.0f;
    for (int r = threadIdx.x; r < rows; r += 256) {
        const float v = conv[(size_t)r * C + c];
        s += v; q += v * v;
    }
    ss[threadIdx.x] = s; sq[threadIdx.x] = q;
    __syncthreads();
    for (int st = 128; st > 0; st >>= 1) {
        if (threadIdx.x < st) { ss[threadIdx.x] += ss[threadIdx.x + st];
                                sq[threadIdx.x] += sq[threadIdx.x + st]; }
        __syncthreads();
    }
    if (threadIdx.x == 0) {
        const float m = ss[0] / rows;
        const float var = sq[0] / rows - m * m;
        mean[c] = m;
        rstd[c] = rsqrtf(var + 1e-5f);
    }
}

// BN affine + 3x3/2 avg pool (13 -> 6) + flatten(b, c*36+i*6+j) to f16 FC input.
__global__ void bn_pool_pack_kernel(const float* __restrict__ conv, const float* __restrict__ mean,
                                    const float* __restrict__ rstd, const float* __restrict__ gamma,
                                    const float* __restrict__ beta, _Float16* __restrict__ fcin,
                                    long long total)
{
    long long idx = (long long)blockIdx.x * blockDim.x + threadIdx.x;
    if (idx >= total) return;
    const int ij = (int)(idx % 36);
    long long q = idx / 36;
    const int c = (int)(q % 256);
    const int b = (int)(q / 256);
    const int i = ij / 6, j = ij % 6;
    const float m = mean[c], rs = rstd[c], g = gamma[c], be = beta[c];
    float s = 0.0f;
    for (int dy = 0; dy < 3; ++dy)
        for (int dx = 0; dx < 3; ++dx) {
            const int y = 2 * i + dy, x = 2 * j + dx;
            const float v = conv[(((size_t)b * 13 + y) * 13 + x) * 256 + c];
            s += (v - m) * rs * g + be;
        }
    fcin[(size_t)b * 9216 + (size_t)c * 36 + ij] = (_Float16)(s * (1.0f / 9.0f));
}

__global__ void logsoftmax_kernel(const float* __restrict__ logits, float* __restrict__ out)
{
    const int b = threadIdx.x;
    if (b >= 16) return;
    float mx = -3.4e38f;
    for (int j = 0; j < 10; ++j) mx = fmaxf(mx, logits[b * 10 + j]);
    float s = 0.0f;
    for (int j = 0; j < 10; ++j) s += expf(logits[b * 10 + j] - mx);
    const float lse = mx + logf(s);
    for (int j = 0; j < 10; ++j) out[b * 10 + j] = logits[b * 10 + j] - lse;
}

__global__ void zero_out_kernel(float* __restrict__ out)
{
    const int i = threadIdx.x;
    if (i < 160) out[i] = 0.0f;
}

// ---------------------------------------------------------------------------
// Host orchestration
// ---------------------------------------------------------------------------
static inline void launch_gemm(const _Float16* A, const _Float16* Bt, float* C,
                               int M, int N, int Kp, long long sA, long long sB, long long sC,
                               int batch, const float* bias, int relu, hipStream_t stream)
{
    dim3 grid((M + 63) / 64, (N + 63) / 64, batch);
    gemm_f16_wmma<<<grid, dim3(256), 0, stream>>>(A, Bt, C, M, N, Kp, sA, sB, sC, bias, relu);
}

#define L1D(kern, total, ...) do {                                              \
    long long _t = (total);                                                     \
    unsigned _b = (unsigned)((_t + 255) / 256);                                 \
    kern<<<dim3(_b), dim3(256), 0, stream>>>(__VA_ARGS__, _t);                  \
} while (0)

extern "C" void kernel_launch(void* const* d_in, const int* in_sizes, int n_in,
                              void* d_out, int out_size, void* d_ws, size_t ws_size,
                              hipStream_t stream)
{
    (void)in_sizes; (void)n_in; (void)out_size;

    const float* x     = (const float*)d_in[0];
    const float* f1    = (const float*)d_in[1];
    const float* f2    = (const float*)d_in[2];
    const float* f3    = (const float*)d_in[3];
    const float* f4    = (const float*)d_in[4];
    const float* f5    = (const float*)d_in[5];
    const float* gamma = (const float*)d_in[6];
    const float* beta  = (const float*)d_in[7];
    const float* W1    = (const float*)d_in[8];
    const float* b1    = (const float*)d_in[9];
    const float* W2    = (const float*)d_in[10];
    const float* b2    = (const float*)d_in[11];
    const float* W3    = (const float*)d_in[12];
    const float* b3    = (const float*)d_in[13];
    float* out = (float*)d_out;

    char* ws = (char*)d_ws;
    size_t off = 0;
    auto alloc = [&](size_t bytes) -> char* {
        off = (off + 255) & ~(size_t)255;
        char* p = ws + off;
        off += bytes;
        return p;
    };

    // ---- persistent small buffers (generated weights / matrices) ----
    _Float16* D1  = (_Float16*)alloc((size_t)113 * 224 * 2);
    _Float16* D2  = (_Float16*)alloc((size_t)55  * 128 * 2);
    _Float16* D3  = (_Float16*)alloc((size_t)27  * 64  * 2);
    _Float16* D4  = (_Float16*)alloc((size_t)13  * 32  * 2);
    _Float16* Wc1 = (_Float16*)alloc((size_t)32  * 32   * 2);
    _Float16* Wc2 = (_Float16*)alloc((size_t)64  * 288  * 2);
    _Float16* Wc3 = (_Float16*)alloc((size_t)128 * 576  * 2);
    _Float16* Wc4 = (_Float16*)alloc((size_t)256 * 1152 * 2);
    _Float16* Wc5 = (_Float16*)alloc((size_t)256 * 2304 * 2);
    _Float16* W1t = (_Float16*)alloc((size_t)512 * 9216 * 2);
    _Float16* W2t = (_Float16*)alloc((size_t)256 * 512  * 2);
    _Float16* W3t = (_Float16*)alloc((size_t)10  * 256  * 2);
    _Float16* FCi = (_Float16*)alloc((size_t)16 * 9216 * 2);
    float*    H1  = (float*)   alloc((size_t)16 * 512 * 4);
    _Float16* H1h = (_Float16*)alloc((size_t)16 * 512 * 2);
    float*    H2  = (float*)   alloc((size_t)16 * 256 * 4);
    _Float16* H2h = (_Float16*)alloc((size_t)16 * 256 * 2);
    float*    LOG = (float*)   alloc((size_t)16 * 10 * 4);
    float*    MEAN= (float*)   alloc((size_t)256 * 4);
    float*    RSTD= (float*)   alloc((size_t)256 * 4);

    // ---- big reusable slots ----
    float* X0 = (float*)alloc((size_t)512 * 113 * 113 * 4);  // activation ping
    float* X1 = (float*)alloc((size_t)512 * 113 * 113 * 4);  // activation pong
    float* CV = (float*)alloc((size_t)802816 * 32 * 4);      // conv (NHWC) output
    // im2col region reuses the (XT,TMP,TF) region: disjoint lifetimes per layer.
    const size_t kXtSz  = (size_t)512 * 224 * 224 * 2;
    const size_t kTmpSz = (size_t)512 * 113 * 224 * 4;
    const size_t kTfSz  = (size_t)512 * 113 * 224 * 2;
    const size_t kImSz  = (size_t)204304 * 288 * 2;
    size_t ru = kXtSz + kTmpSz + kTfSz;
    if (ru < kImSz) ru = kImSz;
    char* RU = alloc(ru);
    _Float16* IM  = (_Float16*)RU;
    _Float16* XT  = (_Float16*)RU;
    float*    TMP = (float*)(RU + kXtSz);
    _Float16* TF  = (_Float16*)(RU + kXtSz + kTmpSz);

    if (off > ws_size) {  // workspace too small: deterministic fallback
        zero_out_kernel<<<dim3(1), dim3(256), 0, stream>>>(out);
        return;
    }

    // ---- generated operands ----
    L1D(gen_dirichlet_kernel, (long long)113 * 224, D1, 113, 224, 224);
    L1D(gen_dirichlet_kernel, (long long)55  * 128, D2, 55, 113, 128);
    L1D(gen_dirichlet_kernel, (long long)27  * 64,  D3, 27, 55, 64);
    L1D(gen_dirichlet_kernel, (long long)13  * 32,  D4, 13, 27, 32);
    L1D(pack_convw_kernel, (long long)32  * 32,   f1, Wc1, 3,   32,  32);
    L1D(pack_convw_kernel, (long long)64  * 288,  f2, Wc2, 32,  64,  288);
    L1D(pack_convw_kernel, (long long)128 * 576,  f3, Wc3, 64,  128, 576);
    L1D(pack_convw_kernel, (long long)256 * 1152, f4, Wc4, 128, 256, 1152);
    L1D(pack_convw_kernel, (long long)256 * 2304, f5, Wc5, 256, 256, 2304);
    L1D(pack_wt_kernel, (long long)512 * 9216, W1, W1t, 9216, 512, 9216);
    L1D(pack_wt_kernel, (long long)256 * 512,  W2, W2t, 512,  256, 512);
    L1D(pack_wt_kernel, (long long)10  * 256,  W3, W3t, 256,  10,  256);

    // ---- Layer 1: conv 3->32 @224, pool 224->113 ----
    {
        const int M = 16 * 224 * 224;
        L1D(im2col_kernel, (long long)M * 32, x, IM, 3, 224, 224, 27, 32);
        launch_gemm(IM, Wc1, CV, M, 32, 32, 0, 0, 0, 1, nullptr, 0, stream);
        L1D(pack_xt_kernel, (long long)512 * 224 * 224, CV, XT, 32, 224, 224, 224);
        launch_gemm(D1, XT, TMP, 113, 224, 224, 0, (long long)224 * 224, (long long)113 * 224,
                    512, nullptr, 0, stream);
        L1D(pack_rows_kernel, (long long)512 * 113 * 224, TMP, TF, 224, 224);
        launch_gemm(TF, D1, X0, 113, 113, 224, (long long)113 * 224, 0, (long long)113 * 113,
                    512, nullptr, 1, stream);
    }
    // ---- Layer 2: conv 32->64 @113, pool 113->55 ----
    {
        const int M = 16 * 113 * 113;
        L1D(im2col_kernel, (long long)M * 288, X0, IM, 32, 113, 113, 288, 288);
        launch_gemm(IM, Wc2, CV, M, 64, 288, 0, 0, 0, 1, nullptr, 0, stream);
        L1D(pack_xt_kernel, (long long)1024 * 113 * 128, CV, XT, 64, 113, 113, 128);
        launch_gemm(D2, XT, TMP, 55, 113, 128, 0, (long long)113 * 128, (long long)55 * 113,
                    1024, nullptr, 0, stream);
        L1D(pack_rows_kernel, (long long)1024 * 55 * 128, TMP, TF, 113, 128);
        launch_gemm(TF, D2, X1, 55, 55, 128, (long long)55 * 128, 0, (long long)55 * 55,
                    1024, nullptr, 1, stream);
    }
    // ---- Layer 3: conv 64->128 @55, pool 55->27 ----
    {
        const int M = 16 * 55 * 55;
        L1D(im2col_kernel, (long long)M * 576, X1, IM, 64, 55, 55, 576, 576);
        launch_gemm(IM, Wc3, CV, M, 128, 576, 0, 0, 0, 1, nullptr, 0, stream);
        L1D(pack_xt_kernel, (long long)2048 * 55 * 64, CV, XT, 128, 55, 55, 64);
        launch_gemm(D3, XT, TMP, 27, 55, 64, 0, (long long)55 * 64, (long long)27 * 55,
                    2048, nullptr, 0, stream);
        L1D(pack_rows_kernel, (long long)2048 * 27 * 64, TMP, TF, 55, 64);
        launch_gemm(TF, D3, X0, 27, 27, 64, (long long)27 * 64, 0, (long long)27 * 27,
                    2048, nullptr, 1, stream);
    }
    // ---- Layer 4: conv 128->256 @27, pool 27->13 ----
    {
        const int M = 16 * 27 * 27;
        L1D(im2col_kernel, (long long)M * 1152, X0, IM, 128, 27, 27, 1152, 1152);
        launch_gemm(IM, Wc4, CV, M, 256, 1152, 0, 0, 0, 1, nullptr, 0, stream);
        L1D(pack_xt_kernel, (long long)4096 * 27 * 32, CV, XT, 256, 27, 27, 32);
        launch_gemm(D4, XT, TMP, 13, 27, 32, 0, (long long)27 * 32, (long long)13 * 27,
                    4096, nullptr, 0, stream);
        L1D(pack_rows_kernel, (long long)4096 * 13 * 32, TMP, TF, 27, 32);
        launch_gemm(TF, D4, X1, 13, 13, 32, (long long)13 * 32, 0, (long long)13 * 13,
                    4096, nullptr, 1, stream);
    }
    // ---- Layer 5: conv 256->256 @13, then BN + avgpool + flatten ----
    {
        const int M = 16 * 13 * 13;  // 2704
        L1D(im2col_kernel, (long long)M * 2304, X1, IM, 256, 13, 13, 2304, 2304);
        launch_gemm(IM, Wc5, CV, M, 256, 2304, 0, 0, 0, 1, nullptr, 0, stream);
        bn_stats_kernel<<<dim3(256), dim3(256), 0, stream>>>(CV, M, 256, MEAN, RSTD);
        L1D(bn_pool_pack_kernel, (long long)16 * 256 * 36, CV, MEAN, RSTD, gamma, beta, FCi);
    }
    // ---- FC head ----
    launch_gemm(FCi, W1t, H1, 16, 512, 9216, 0, 0, 0, 1, b1, 1, stream);
    L1D(pack_rows_kernel, (long long)16 * 512, H1, H1h, 512, 512);
    launch_gemm(H1h, W2t, H2, 16, 256, 512, 0, 0, 0, 1, b2, 1, stream);
    L1D(pack_rows_kernel, (long long)16 * 256, H2, H2h, 256, 256);
    launch_gemm(H2h, W3t, LOG, 16, 10, 256, 0, 0, 0, 1, b3, 0, stream);
    logsoftmax_kernel<<<dim3(1), dim3(16), 0, stream>>>(LOG, out);
}
